// BinarySelfAttention_66640712564849
// MI455X (gfx1250) — matrix operand order repaired and checked by reading, not compiled, not run on hardware
//
#include <hip/hip_runtime.h>
#include <hip/hip_bf16.h>
#include <math.h>

typedef __bf16 bf16;
typedef __attribute__((ext_vector_type(8)))  bf16  bf16x8;
typedef __attribute__((ext_vector_type(16))) bf16  bf16x16;
typedef __attribute__((ext_vector_type(8)))  float f32x8;

#define B_    2
#define T_    2048
#define DM    1024
#define NH    16
#define HD    64
#define MROWS (B_ * T_)   // 4096

// ---------------------------------------------------------------------------
// CDNA5 async global->LDS copy (ASYNCcnt-tracked, bypasses VGPRs)
// ---------------------------------------------------------------------------
__device__ __forceinline__ void async_ld_b128(bf16* lds_ptr, const bf16* gptr) {
  const unsigned lds_off = (unsigned)(unsigned long long)lds_ptr; // LDS offset = addr[31:0]
  asm volatile("global_load_async_to_lds_b128 %0, %1, off"
               :: "v"(lds_off), "v"(gptr)
               : "memory");
}
#define WAIT_ASYNCCNT(n) asm volatile("s_wait_asynccnt " #n ::: "memory")

// ---------------------------------------------------------------------------
// WMMA helpers (CDNA5 wave32, 16x16x32 bf16 -> f32)
// ---------------------------------------------------------------------------
__device__ __forceinline__ f32x8 wmma_bf16(bf16x16 a, bf16x16 b, f32x8 c) {
  return __builtin_amdgcn_wmma_f32_16x16x32_bf16(
      /*neg_a=*/false, a, /*neg_b=*/false, b,
      /*c_mod=*/(short)0, c, /*reuse_a=*/false, /*reuse_b=*/false);
}

// A fragment: lane(r,h) holds row r, K = {8h..8h+7} U {16+8h..16+8h+7}
__device__ __forceinline__ bf16x16 ld_frag_a(const bf16* base, int stride) {
  const int lane = threadIdx.x & 31;
  const int r = lane & 15, h = lane >> 4;
  const bf16* p = base + (size_t)r * stride + h * 8;
  bf16x8 lo = *(const bf16x8*)p;
  bf16x8 hi = *(const bf16x8*)(p + 16);
  bf16x16 f;
#pragma unroll
  for (int i = 0; i < 8; ++i) { f[i] = lo[i]; f[i + 8] = hi[i]; }
  return f;
}

// B fragment: lane(r,h) holds column r, K = {8h..8h+7} U {16+8h..16+8h+7}
__device__ __forceinline__ bf16x16 ld_frag_b(const bf16* base, int stride) {
  const int lane = threadIdx.x & 31;
  const int r = lane & 15, h = lane >> 4;
  const bf16* p = base + (size_t)r * stride + h * 8;
  bf16x8 lo = *(const bf16x8*)p;
  bf16x8 hi = *(const bf16x8*)(p + 16);
  bf16x16 f;
#pragma unroll
  for (int i = 0; i < 8; ++i) { f[i] = lo[i]; f[i + 8] = hi[i]; }
  return f;
}

// ---------------------------------------------------------------------------
// Ternary quantization: fixed-point abs-mean (deterministic u64 atomics)
// ---------------------------------------------------------------------------
__global__ void init_acc_kernel(unsigned long long* acc) {
  if (threadIdx.x < 4) acc[threadIdx.x] = 0ull;
}

__global__ __launch_bounds__(256) void absmean_kernel(
    const float* __restrict__ W0, const float* __restrict__ W1,
    const float* __restrict__ W2, const float* __restrict__ W3,
    unsigned long long* __restrict__ acc) {
  const int w = blockIdx.x >> 10;           // 1024 blocks per 1Mi-element matrix
  const int blk = blockIdx.x & 1023;
  const float* W = (w == 0) ? W0 : (w == 1) ? W1 : (w == 2) ? W2 : W3;
  float s = 0.f;
  const int base = blk * 1024 + threadIdx.x;
#pragma unroll
  for (int i = 0; i < 4; ++i) s += fabsf(W[base + i * 256]);
#pragma unroll
  for (int off = 16; off > 0; off >>= 1) s += __shfl_down(s, off, 32);
  __shared__ float ws[8];
  if ((threadIdx.x & 31) == 0) ws[threadIdx.x >> 5] = s;
  __syncthreads();
  if (threadIdx.x == 0) {
    float t = 0.f;
#pragma unroll
    for (int i = 0; i < 8; ++i) t += ws[i];
    atomicAdd(&acc[w], (unsigned long long)((double)t * 1048576.0));  // *2^20
  }
}

__global__ __launch_bounds__(256) void quant_kernel(
    const float* __restrict__ W0, const float* __restrict__ W1,
    const float* __restrict__ W2, const float* __restrict__ W3,
    const unsigned long long* __restrict__ acc, bf16* __restrict__ Wt) {
  const unsigned idx = blockIdx.x * 256u + threadIdx.x;   // 0 .. 4Mi-1
  const int w = idx >> 20;
  const int e = idx & 0xFFFFF;
  const float* W = (w == 0) ? W0 : (w == 1) ? W1 : (w == 2) ? W2 : W3;
  const float scale = (float)((double)acc[w] / 1048576.0 / 1048576.0);
  float v = W[e] / (scale + 1e-8f);
  v = fminf(1.f, fmaxf(-1.f, v));
  Wt[idx] = (bf16)(rintf(v) * scale);   // rintf = round-half-even, matches jnp.round
}

__global__ __launch_bounds__(256) void xconv_kernel(const float* __restrict__ x,
                                                    bf16* __restrict__ xb) {
  const unsigned idx = blockIdx.x * 256u + threadIdx.x;
  xb[idx] = (bf16)x[idx];
}

// ---------------------------------------------------------------------------
// GEMM: C[M,N] (f32) = A[M,K] (bf16) * Bt[N,K]^T (bf16).
// Block tile 128x128, K-tile 64, 8 waves each 32x64 (2x4 WMMA frags, 16
// WMMA / stage). Double-buffered LDS fed by async global->LDS copies.
// ---------------------------------------------------------------------------
#define GS   72                 // padded LDS row stride (bf16), rows 144B
#define GTA  (128 * GS)         // one A/B buffer (9216 bf16 = 18KB)

__device__ __forceinline__ void gemm_stage_async(
    const bf16* __restrict__ A, const bf16* __restrict__ Bt,
    bf16* As, bf16* Bs, int row_a0, int row_b0, int K, int k0, int tid) {
#pragma unroll
  for (int i = 0; i < 4; ++i) {                 // 128x64 tile: 4 x 16B / thread
    const int lin = tid + 256 * i;
    const int rr = lin >> 3, cc = (lin & 7) * 8;
    async_ld_b128(As + rr * GS + cc, A  + (size_t)(row_a0 + rr) * K + k0 + cc);
    async_ld_b128(Bs + rr * GS + cc, Bt + (size_t)(row_b0 + rr) * K + k0 + cc);
  }
}

__global__ __launch_bounds__(256) void gemm_bf16_kernel(
    const bf16* __restrict__ A, const bf16* __restrict__ Bt,
    float* __restrict__ C, int M, int N, int K) {
  __shared__ __align__(16) bf16 As[2 * GTA];    // 36 KB
  __shared__ __align__(16) bf16 Bs[2 * GTA];    // 36 KB

  const int tid = threadIdx.x;
  const int wave = tid >> 5;
  const int wm = wave >> 1, wn = wave & 1;      // 4x2 wave grid, 32x64 each
  const int lane = tid & 31;
  const int r = lane & 15, h = lane >> 4;

  const int row_a0 = blockIdx.x * 128;
  const int row_b0 = blockIdx.y * 128;

  f32x8 acc[2][4];
#pragma unroll
  for (int m = 0; m < 2; ++m)
#pragma unroll
    for (int c = 0; c < 4; ++c) acc[m][c] = {};

  const int NT = K >> 6;                        // 64-wide K tiles
  gemm_stage_async(A, Bt, As, Bs, row_a0, row_b0, K, 0, tid);

  for (int kt = 0; kt < NT; ++kt) {
    bf16* Ac = As + (kt & 1) * GTA;
    bf16* Bc = Bs + (kt & 1) * GTA;
    if (kt + 1 < NT) {
      gemm_stage_async(A, Bt, As + ((kt + 1) & 1) * GTA, Bs + ((kt + 1) & 1) * GTA,
                       row_a0, row_b0, K, (kt + 1) * 64, tid);
      if (kt + 2 < NT) {     // L2 prefetch two tiles ahead
        __builtin_prefetch(A  + (size_t)(row_a0 + (tid >> 1)) * K + (kt + 2) * 64, 0, 1);
        __builtin_prefetch(Bt + (size_t)(row_b0 + (tid >> 1)) * K + (kt + 2) * 64, 0, 1);
      }
      WAIT_ASYNCCNT(8);      // 8 newly issued may remain; current tile landed
    } else {
      WAIT_ASYNCCNT(0);
    }
    __syncthreads();

#pragma unroll
    for (int s = 0; s < 2; ++s) {               // two 32-K substeps
      bf16x16 a0 = ld_frag_a(Ac + (wm * 32) * GS + 32 * s, GS);
      bf16x16 a1 = ld_frag_a(Ac + (wm * 32 + 16) * GS + 32 * s, GS);
#pragma unroll
      for (int c = 0; c < 4; ++c) {
        bf16x16 b = ld_frag_b(Bc + (wn * 64 + 16 * c) * GS + 32 * s, GS);
        acc[0][c] = wmma_bf16(a0, b, acc[0][c]);
        acc[1][c] = wmma_bf16(a1, b, acc[1][c]);
      }
    }
    __syncthreads();
  }

  const int gm = row_a0 + wm * 32, gn = row_b0 + wn * 64;
#pragma unroll
  for (int m = 0; m < 2; ++m)
#pragma unroll
    for (int c = 0; c < 4; ++c)
#pragma unroll
      for (int i = 0; i < 8; ++i)
        C[(size_t)(gm + m * 16 + 8 * h + i) * N + gn + c * 16 + r] = acc[m][c][i];
}

// ---------------------------------------------------------------------------
// RoPE + pack fp32 [B,T,H*D] -> bf16 [B,H,T,D] (q,k rotated; v passthrough)
// ---------------------------------------------------------------------------
__global__ __launch_bounds__(256) void rope_pack_kernel(
    const float* __restrict__ q, const float* __restrict__ k,
    const float* __restrict__ v, bf16* __restrict__ qb,
    bf16* __restrict__ kb, bf16* __restrict__ vb) {
  const unsigned idx = blockIdx.x * 256u + threadIdx.x;  // over B*T*H*D = 4Mi
  const int d = idx & 63;
  const int hh = (idx >> 6) & 15;
  const int t = (idx >> 10) & 2047;
  const int b = idx >> 21;
  const size_t src = ((size_t)(b * T_ + t)) * DM + hh * HD + d;
  const size_t dst = (((size_t)(b * NH + hh)) * T_ + t) * HD + d;
  const int i = d & 31;
  // inv_freq = 10000^(-i/32) ; log(10000)/32 = 0.28782313662425572
  const float ang = (float)t * expf(-(float)i * 0.28782313662425572f);
  const float c = cosf(ang), s = sinf(ang);
  const size_t partner = src + (d < 32 ? 32 : -32);
  const float qv = q[src], kv = k[src];
  const float qr = (d < 32) ? -q[partner] : q[partner];
  const float kr = (d < 32) ? -k[partner] : k[partner];
  qb[dst] = (bf16)(qv * c + qr * s);
  kb[dst] = (bf16)(kv * c + kr * s);
  vb[dst] = (bf16)v[src];
}

// ---------------------------------------------------------------------------
// Flash-style causal attention. Block = (b, head, 128 q rows); 8 waves x 16
// rows. Keys in tiles of 32; K tile async-staged row-major, V transposed in
// LDS; P bounced through per-wave LDS to re-form A fragments for PV.
// ---------------------------------------------------------------------------
__global__ __launch_bounds__(256) void attn_kernel(
    const bf16* __restrict__ qb, const bf16* __restrict__ kb,
    const bf16* __restrict__ vb, bf16* __restrict__ yb) {
  __shared__ __align__(16) bf16 Kt[32 * 72];       // [j][d], stride 72
  __shared__ __align__(16) bf16 Vt[64 * 40];       // [d][j], stride 40
  __shared__ __align__(16) bf16 Pw[8 * 16 * 40];   // per-wave 16x32, stride 40

  const int tid = threadIdx.x;
  const int lane = tid & 31;
  const int r = lane & 15, h = lane >> 4;
  const int wave = tid >> 5;

  const int qt0 = blockIdx.x * 128;
  const int head = blockIdx.y;
  const int b = blockIdx.z;

  const bf16* qh = qb + ((size_t)(b * NH + head)) * T_ * HD;
  const bf16* kh = kb + ((size_t)(b * NH + head)) * T_ * HD;
  const bf16* vh = vb + ((size_t)(b * NH + head)) * T_ * HD;

  const int qrow0 = qt0 + wave * 16;
  bf16x16 qa0 = ld_frag_a(qh + (size_t)qrow0 * HD + 0, HD);    // d 0..31
  bf16x16 qa1 = ld_frag_a(qh + (size_t)qrow0 * HD + 32, HD);   // d 32..63

  f32x8 o0 = {}, o1 = {}, o2 = {}, o3 = {};
  float mrow[8], lrow[8];
#pragma unroll
  for (int i = 0; i < 8; ++i) { mrow[i] = -__builtin_inff(); lrow[i] = 0.f; }

  bf16* Pme = Pw + wave * 16 * 40;

  for (int j0 = 0; j0 < qt0 + 128; j0 += 32) {
    {   // K tile via async copy engine; V loaded + transposed via VALU path
      const int kr = tid >> 3, kc = (tid & 7) * 8;
      async_ld_b128(Kt + kr * 72 + kc, kh + (size_t)(j0 + kr) * HD + kc);
      bf16x8 vv = *(const bf16x8*)(vh + (size_t)(j0 + kr) * HD + kc);
#pragma unroll
      for (int e = 0; e < 8; ++e) Vt[(kc + e) * 40 + kr] = vv[e];
      WAIT_ASYNCCNT(0);
    }
    __syncthreads();

    // S = Q K^T for 32 keys (two 16-col frags, d reduced in 2 WMMA steps)
    f32x8 s0 = {}, s1 = {};
    s0 = wmma_bf16(qa0, ld_frag_b(Kt + 0 * 72 + 0, 72), s0);
    s0 = wmma_bf16(qa1, ld_frag_b(Kt + 0 * 72 + 32, 72), s0);
    s1 = wmma_bf16(qa0, ld_frag_b(Kt + 16 * 72 + 0, 72), s1);
    s1 = wmma_bf16(qa1, ld_frag_b(Kt + 16 * 72 + 32, 72), s1);

    // scale + causal mask + online softmax (row = 8h+i, col = r / 16+r)
#pragma unroll
    for (int i = 0; i < 8; ++i) {
      const int qg = qrow0 + 8 * h + i;
      float v0 = s0[i] * 0.125f, v1 = s1[i] * 0.125f;
      if (j0 + r > qg)      v0 = -__builtin_inff();
      if (j0 + 16 + r > qg) v1 = -__builtin_inff();
      float t = fmaxf(v0, v1);
#pragma unroll
      for (int off = 8; off > 0; off >>= 1)
        t = fmaxf(t, __shfl_xor(t, off, 32));      // stays within 16-lane half
      const float mnew = fmaxf(mrow[i], t);
      const float alpha = expf(mrow[i] - mnew);
      const float e0 = expf(v0 - mnew), e1 = expf(v1 - mnew);
      float sum = e0 + e1;
#pragma unroll
      for (int off = 8; off > 0; off >>= 1)
        sum += __shfl_xor(sum, off, 32);
      lrow[i] = lrow[i] * alpha + sum;
      mrow[i] = mnew;
      o0[i] *= alpha; o1[i] *= alpha; o2[i] *= alpha; o3[i] *= alpha;
      Pme[(8 * h + i) * 40 + r] = (bf16)e0;
      Pme[(8 * h + i) * 40 + 16 + r] = (bf16)e1;
    }
    __syncthreads();   // make P visible for A-fragment reload

    bf16x16 pa = ld_frag_a(Pme, 40);
    o0 = wmma_bf16(pa, ld_frag_b(Vt + 0 * 40, 40), o0);
    o1 = wmma_bf16(pa, ld_frag_b(Vt + 16 * 40, 40), o1);
    o2 = wmma_bf16(pa, ld_frag_b(Vt + 32 * 40, 40), o2);
    o3 = wmma_bf16(pa, ld_frag_b(Vt + 48 * 40, 40), o3);
    __syncthreads();   // protect Kt/Vt before next stage
  }

#pragma unroll
  for (int i = 0; i < 8; ++i) {
    const int qg = qrow0 + 8 * h + i;
    const float inv = 1.f / lrow[i];
    const size_t row = (size_t)(b * T_ + qg) * DM + head * HD;
    yb[row + 0 * 16 + r] = (bf16)(o0[i] * inv);
    yb[row + 1 * 16 + r] = (bf16)(o1[i] * inv);
    yb[row + 2 * 16 + r] = (bf16)(o2[i] * inv);
    yb[row + 3 * 16 + r] = (bf16)(o3[i] * inv);
  }
}

// ---------------------------------------------------------------------------
extern "C" void kernel_launch(void* const* d_in, const int* in_sizes, int n_in,
                              void* d_out, int out_size, void* d_ws, size_t ws_size,
                              hipStream_t stream) {
  const float* x  = (const float*)d_in[0];
  const float* Wq = (const float*)d_in[1];
  const float* Wk = (const float*)d_in[2];
  const float* Wv = (const float*)d_in[3];
  const float* Wo = (const float*)d_in[4];
  // d_in[5] = causal mask: handled analytically in attn_kernel

  const size_t need = 256 + ((size_t)88 << 20);
  if (ws_size < need) return;

  char* ws = (char*)d_ws;
  unsigned long long* acc = (unsigned long long*)ws;               // 32 B
  bf16*  Wt = (bf16*)(ws + 256);                                   // 8 MiB
  bf16*  xb = (bf16*)(ws + 256 + ((size_t)8 << 20));               // 8 MiB
  float* qf = (float*)(ws + 256 + ((size_t)16 << 20));             // 16 MiB
  float* kf = qf + (size_t)MROWS * DM;                             // 16 MiB
  float* vf = kf + (size_t)MROWS * DM;                             // 16 MiB
  bf16*  qb = (bf16*)(ws + 256 + ((size_t)64 << 20));              // 8 MiB
  bf16*  kb = qb + (size_t)B_ * NH * T_ * HD;                      // 8 MiB
  bf16*  vb = kb + (size_t)B_ * NH * T_ * HD;                      // 8 MiB
  bf16*  yb = (bf16*)qf;   // reuse q fp32 region (dead after rope_pack)

  init_acc_kernel<<<1, 32, 0, stream>>>(acc);
  absmean_kernel<<<4096, 256, 0, stream>>>(Wq, Wk, Wv, Wo, acc);
  quant_kernel<<<16384, 256, 0, stream>>>(Wq, Wk, Wv, Wo, acc, Wt);
  xconv_kernel<<<16384, 256, 0, stream>>>(x, xb);

  dim3 ggrid(MROWS / 128, DM / 128);
  gemm_bf16_kernel<<<ggrid, 256, 0, stream>>>(xb, Wt + 0 * (1u << 20), qf, MROWS, DM, DM);
  gemm_bf16_kernel<<<ggrid, 256, 0, stream>>>(xb, Wt + 1 * (1u << 20), kf, MROWS, DM, DM);
  gemm_bf16_kernel<<<ggrid, 256, 0, stream>>>(xb, Wt + 2 * (1u << 20), vf, MROWS, DM, DM);

  rope_pack_kernel<<<16384, 256, 0, stream>>>(qf, kf, vf, qb, kb, vb);

  attn_kernel<<<dim3(T_ / 128, NH, B_), 256, 0, stream>>>(qb, kb, vb, yb);

  gemm_bf16_kernel<<<ggrid, 256, 0, stream>>>(yb, Wt + 3 * (1u << 20),
                                              (float*)d_out, MROWS, DM, DM);
}
